// ModulatedSparseTransformerCrossBlock_41747082117385
// MI455X (gfx1250) — matrix operand order repaired and checked by reading, not compiled, not run on hardware
//
#include <hip/hip_runtime.h>
#include <hip/hip_bf16.h>

// ---------------- types ----------------
typedef __attribute__((ext_vector_type(16))) __bf16 v16bf;
typedef __attribute__((ext_vector_type(8)))  __bf16 v8bf;
typedef __attribute__((ext_vector_type(8)))  float  v8f;
typedef __attribute__((ext_vector_type(4)))  unsigned int v4u;
typedef __attribute__((ext_vector_type(4)))  int v4i;
typedef __attribute__((ext_vector_type(8)))  int v8i;

union BF16x16 { v16bf v; v8bf h[2]; __bf16 e[16]; };

// TDM availability: device pass only (host pass falls back; kernel ABI unchanged)
#if defined(__gfx1250__) && __has_builtin(__builtin_amdgcn_tensor_load_to_lds)
#define HAVE_TDM 1
#if __clang_major__ >= 23
#define TDM_CALL(g0, g1, g2, g3) \
  __builtin_amdgcn_tensor_load_to_lds(g0, g1, g2, g3, (v8i){0, 0, 0, 0, 0, 0, 0, 0}, 0)
#else
#define TDM_CALL(g0, g1, g2, g3) __builtin_amdgcn_tensor_load_to_lds(g0, g1, g2, g3, 0)
#endif
#else
#define HAVE_TDM 0
#endif

__device__ __forceinline__ v8f wmma_bf16(v16bf a, v16bf b, v8f c) {
  return __builtin_amdgcn_wmma_f32_16x16x32_bf16(false, a, false, b, (short)0, c, false, false);
}

// A-fragment (16x32 bf16). `p` = tile_base + (lane&15)*ld (this lane's row, K=0).
// Element e <-> K = e + 16*(lane>>4).
__device__ __forceinline__ v16bf load_a_ptr(const __bf16* p) {
  int lane = threadIdx.x & 31;
  int lh = lane >> 4;
  BF16x16 u;
  u.h[0] = *(const v8bf*)(p + 16 * lh);      // K = 16h .. 16h+7
  u.h[1] = *(const v8bf*)(p + 16 * lh + 8);  // K = 16h+8 .. 16h+15
  return u.v;
}

// B-fragment (32x16 bf16) from K-major tile (rows = K, row length ld).
// `base` already includes the per-lane column (lane&15). Element e <-> K = e + 16*(lane>>4).
__device__ __forceinline__ v16bf load_b_strided(const __bf16* base, int ld) {
  int lane = threadIdx.x & 31;
  int lh = lane >> 4;
  BF16x16 u;
#pragma unroll
  for (int e = 0; e < 16; e++) u.e[e] = base[(size_t)(e + 16 * lh) * ld];
  return u.v;
}

__device__ __forceinline__ float dev_silu(float x) { return x / (1.f + __expf(-x)); }

__device__ __forceinline__ unsigned lds_addr_of(const void* p) {
  return (unsigned)(unsigned long long)(__attribute__((address_space(3))) const char*)p;
}

#if HAVE_TDM
// Issue one 2-D TDM tile load: global (row-major, rowStride elems of bf16) -> LDS
// (contiguous tileW x tileH). D# per CDNA5 ISA 8.3/8.4.
__device__ __forceinline__ void tdm_load_2d(unsigned ldsOff, const void* gaddr,
                                            unsigned tileW, unsigned tileH,
                                            unsigned rowStride) {
  unsigned long long ga = (unsigned long long)gaddr;
  v4u g0;
  g0[0] = 1u;                                               // count=1, user mode
  g0[1] = ldsOff;                                           // lds_addr (bytes)
  g0[2] = (unsigned)(ga & 0xffffffffu);                     // global_addr[31:0]
  g0[3] = (unsigned)((ga >> 32) & 0x01ffffffu) | 0x80000000u; // ga[56:32] | type=2
  v8i g1;
  g1[0] = (int)(1u << 16);                                  // data_size=1 (2 bytes)
  g1[1] = (int)((tileW & 0xffffu) << 16);                   // tensor_dim0[15:0]
  g1[2] = (int)((tileW >> 16) | ((tileH & 0xffffu) << 16)); // dim0 hi | tensor_dim1 lo
  g1[3] = (int)((tileH >> 16) | (tileW << 16));             // dim1 hi | tile_dim0
  g1[4] = (int)(tileH & 0xffffu);                           // tile_dim1 (tile_dim2=0)
  g1[5] = (int)rowStride;                                   // tensor_dim0_stride[31:0]
  g1[6] = 0;                                                // stride hi, dim1_stride lo
  g1[7] = 0;
  v4i z = {0, 0, 0, 0};
  TDM_CALL(g0, g1, z, z);
}
#endif

// ---------------- constants ----------------
#define CC   1024
#define HH   16
#define DD   64
#define WINW 512
#define SHIFTW 256
#define TOPKK 16
#define HIDN 4096
#define LL   4096

// ---------------- m = silu(mod) @ w_ada + b_ada ----------------
__global__ __launch_bounds__(128) void mod_gemm_kernel(
    const float* __restrict__ mod, const float* __restrict__ w,
    const float* __restrict__ bias, float* __restrict__ m) {
  __shared__ float sm[CC];
  int b = blockIdx.y;
  int col = blockIdx.x * 128 + threadIdx.x;
  for (int k = threadIdx.x; k < CC; k += 128) {
    float v = mod[b * CC + k];
    sm[k] = v / (1.f + __expf(-v));
  }
  __syncthreads();
  float acc = bias[col];
  for (int k = 0; k < CC; k++) acc += sm[k] * w[(size_t)k * (9 * CC) + col];
  m[b * (9 * CC) + col] = acc;
}

// ---------------- f32 -> bf16 convert ----------------
__global__ __launch_bounds__(256) void cvt_f32_bf16(const float* __restrict__ in,
                                                    __bf16* __restrict__ out, int n) {
  for (int i = blockIdx.x * 256 + threadIdx.x; i < n; i += gridDim.x * 256)
    out[i] = (__bf16)in[i];
}

// ---------------- fused LayerNorm + modulation -> bf16 ----------------
__global__ __launch_bounds__(256) void ln_mod_kernel(
    const float* __restrict__ x, const float* __restrict__ mbuf,
    int shChunk, int scChunk, __bf16* __restrict__ out, int Lrows) {
  int row = blockIdx.x;
  int b = row / Lrows;
  const float* xr = x + (size_t)row * CC;
  float v[4], s = 0.f, sq = 0.f;
#pragma unroll
  for (int i = 0; i < 4; i++) {
    v[i] = xr[threadIdx.x + 256 * i];
    s += v[i];
    sq += v[i] * v[i];
  }
#pragma unroll
  for (int off = 16; off; off >>= 1) {
    s  += __shfl_down(s, off, 32);
    sq += __shfl_down(sq, off, 32);
  }
  __shared__ float rs[8], rq[8], stats[2];
  int wid = threadIdx.x >> 5;
  if ((threadIdx.x & 31) == 0) { rs[wid] = s; rq[wid] = sq; }
  __syncthreads();
  if (threadIdx.x == 0) {
    float ts = 0.f, tq = 0.f;
    for (int i = 0; i < 8; i++) { ts += rs[i]; tq += rq[i]; }
    float mean = ts * (1.f / CC);
    float var = tq * (1.f / CC) - mean * mean;
    stats[0] = mean;
    stats[1] = rsqrtf(var + 1e-6f);
  }
  __syncthreads();
  float mean = stats[0], rstd = stats[1];
  const float* sh = mbuf + b * (9 * CC) + shChunk * CC;
  const float* sc = mbuf + b * (9 * CC) + scChunk * CC;
  __bf16* orow = out + (size_t)row * CC;
#pragma unroll
  for (int i = 0; i < 4; i++) {
    int c = threadIdx.x + 256 * i;
    float hn = (v[i] - mean) * rstd;
    orow[c] = (__bf16)(hn * (1.f + sc[c]) + sh[c]);
  }
}

// ---------------- generic WMMA bf16 GEMM ----------------
// MODE 0: out_bf16 = A@B + bias                      (block 128x128, wave 64x32)
// MODE 1: out_f32  = res + gate[b][c]*(A@B + bias)   (block 128x128, wave 64x32)
// MODE 2: out_bf16 = silu(A@B1+b1)*(A@B2+b2)         (block 128x64,  wave 32x32, B2=cols[HID,2HID))
template <int MODE>
__global__ __launch_bounds__(256) void gemm_bf16_kernel(
    const __bf16* __restrict__ A, const __bf16* __restrict__ Bw,
    const float* __restrict__ bias, __bf16* __restrict__ outB,
    float* __restrict__ outF, const float* __restrict__ res,
    const float* __restrict__ mbuf, int gateChunk,
    int M, int N, int K, int ldb, int Lrows) {
  constexpr int BNT = (MODE == 2) ? 64 : 128;  // block N extent
  constexpr int WMT = (MODE == 2) ? 2 : 4;     // 16-row tiles per wave
  constexpr int NB  = (MODE == 2) ? 3 : 2;     // TDM ops per stage batch

  __shared__ __bf16 As[2][128 * 32];
  __shared__ __bf16 Bs[2][32 * BNT];
  __shared__ __bf16 Bs2[2][(MODE == 2) ? (32 * 64) : 32];

  int tid = threadIdx.x, lane = tid & 31, wv = tid >> 5;
  int waveM = (MODE == 2) ? (wv >> 1) : (wv >> 2);
  int waveN = (MODE == 2) ? (wv & 1) : (wv & 3);
  int lh = lane >> 4, m16 = lane & 15;
  int bM = blockIdx.y * 128, bN = blockIdx.x * BNT;

  v8f acc[WMT][2];
  v8f acc2[WMT][2];
#pragma unroll
  for (int i = 0; i < WMT; i++) {
    acc[i][0] = {}; acc[i][1] = {};
    acc2[i][0] = {}; acc2[i][1] = {};
  }

  auto mma_step = [&](const __bf16* Asb, const __bf16* Bsb, const __bf16* Bs2b) {
    v16bf a[WMT];
#pragma unroll
    for (int i = 0; i < WMT; i++)
      a[i] = load_a_ptr(Asb + (size_t)(waveM * WMT * 16 + i * 16 + m16) * 32);
#pragma unroll
    for (int jn = 0; jn < 2; jn++) {
      v16bf b = load_b_strided(Bsb + waveN * 32 + jn * 16 + m16, BNT);
#pragma unroll
      for (int i = 0; i < WMT; i++) acc[i][jn] = wmma_bf16(a[i], b, acc[i][jn]);
      if (MODE == 2) {
        v16bf b2 = load_b_strided(Bs2b + waveN * 32 + jn * 16 + m16, 64);
#pragma unroll
        for (int i = 0; i < WMT; i++) acc2[i][jn] = wmma_bf16(a[i], b2, acc2[i][jn]);
      }
    }
  };

  int nk = K / 32;

#if HAVE_TDM
  auto tdm_stage = [&](int buf, int k0) {
    tdm_load_2d(lds_addr_of(&As[buf][0]), A + (size_t)bM * K + k0, 32, 128, K);
    tdm_load_2d(lds_addr_of(&Bs[buf][0]), Bw + (size_t)k0 * ldb + bN, BNT, 32, ldb);
    if (MODE == 2)
      tdm_load_2d(lds_addr_of(&Bs2[buf][0]), Bw + (size_t)k0 * ldb + bN + HIDN, 64, 32, ldb);
  };
  if (wv == 0) tdm_stage(0, 0);
  for (int kt = 0; kt < nk; kt++) {
    int cur = kt & 1;
    if (wv == 0) {
      if (kt + 1 < nk) {
        tdm_stage(cur ^ 1, (kt + 1) * 32);
        __builtin_amdgcn_s_wait_tensorcnt(NB);  // previous batch (cur) complete
      } else {
        __builtin_amdgcn_s_wait_tensorcnt(0);
      }
    }
    __syncthreads();
    mma_step(As[cur], Bs[cur], Bs2[cur]);
    __syncthreads();
  }
#else
  int arow = tid >> 1, aseg = tid & 1;
  for (int kt = 0; kt < nk; kt++) {
    int k0 = kt * 32;
    __syncthreads();
    *(v16bf*)(As[0] + arow * 32 + aseg * 16) =
        *(const v16bf*)(A + (size_t)(bM + arow) * K + k0 + aseg * 16);
    if (MODE == 2) {
      int brow = (tid & 127) >> 2, bseg = tid & 3;
      const __bf16* src = Bw + (size_t)(k0 + brow) * ldb + bN +
                          ((tid >= 128) ? HIDN : 0) + bseg * 16;
      __bf16* dst = ((tid >= 128) ? Bs2[0] : Bs[0]) + brow * 64 + bseg * 16;
      *(v16bf*)dst = *(const v16bf*)src;
    } else {
      int brow = tid >> 3, bseg = tid & 7;
      *(v16bf*)(Bs[0] + brow * BNT + bseg * 16) =
          *(const v16bf*)(Bw + (size_t)(k0 + brow) * ldb + bN + bseg * 16);
    }
    __syncthreads();
    mma_step(As[0], Bs[0], Bs2[0]);
  }
#endif

#pragma unroll
  for (int i = 0; i < WMT; i++)
#pragma unroll
    for (int jn = 0; jn < 2; jn++) {
#pragma unroll
      for (int j = 0; j < 8; j++) {
        int r = bM + waveM * WMT * 16 + i * 16 + j + 8 * lh;
        int c = bN + waveN * 32 + jn * 16 + m16;
        float v = acc[i][jn][j] + bias[c];
        if (MODE == 0) {
          outB[(size_t)r * N + c] = (__bf16)v;
        } else if (MODE == 1) {
          int bb = r / Lrows;
          float g = mbuf[bb * (9 * CC) + gateChunk * CC + c];
          outF[(size_t)r * N + c] = res[(size_t)r * N + c] + g * v;
        } else {
          float v2 = acc2[i][jn][j] + bias[c + HIDN];
          outB[(size_t)r * N + c] = (__bf16)(dev_silu(v) * v2);
        }
      }
    }
}

// ---------------- shifted-window self-attention ----------------
// qkv: (B, L, 3*C) bf16, [3][H][D] per row. Roll(-256) folded into r=(w*512+pos+256)&4095
// for q/k/v loads AND the o store (rolls back). block=(b,w,h), 64 thr = 2 waves,
// dyn LDS = Vt[64][512] bf16 + 2x S[512][16] f32.
__global__ __launch_bounds__(64) void self_attn_kernel(
    const __bf16* __restrict__ qkv, __bf16* __restrict__ o) {
  extern __shared__ char sa_smem[];
  __bf16* Vt = (__bf16*)sa_smem;                  // [64][512]
  float* Sb = (float*)(sa_smem + DD * WINW * 2);  // 2 x [512][16]
  int blk = blockIdx.x;
  int h = blk & 15, w = (blk >> 4) & 7, b = blk >> 7;
  int tid = threadIdx.x, lane = tid & 31, wv = tid >> 5;
  int lh = lane >> 4, m16 = lane & 15;
  float* S = Sb + wv * WINW * 16;
  size_t base = (size_t)b * LL;

  for (int it = 0; it < 8; it++) {
    int key = it * 64 + tid;
    int r = (w * WINW + key + SHIFTW) & (LL - 1);
    const __bf16* vp = qkv + (base + r) * (3 * CC) + 2 * CC + h * DD;
#pragma unroll
    for (int c4 = 0; c4 < 4; c4++) {
      BF16x16 u;
      u.v = *(const v16bf*)(vp + c4 * 16);
#pragma unroll
      for (int e = 0; e < 16; e++) Vt[(size_t)(c4 * 16 + e) * WINW + key] = u.e[e];
    }
  }
  __syncthreads();

  for (int qt = wv; qt < WINW / 16; qt += 2) {
    int qr = (w * WINW + qt * 16 + m16 + SHIFTW) & (LL - 1);
    const __bf16* qp = qkv + (base + qr) * (3 * CC) + h * DD;
    v16bf aQ0 = load_a_ptr(qp);
    v16bf aQ1 = load_a_ptr(qp + 32);

    for (int kt = 0; kt < WINW / 16; kt++) {
      int rk = (w * WINW + kt * 16 + m16 + SHIFTW) & (LL - 1);
      const __bf16* kp = qkv + (base + rk) * (3 * CC) + CC + h * DD;
      v16bf bK0 = *(const v16bf*)(kp + 16 * lh);
      v16bf bK1 = *(const v16bf*)(kp + 32 + 16 * lh);
      v8f s = {};
      s = wmma_bf16(aQ0, bK0, s);
      s = wmma_bf16(aQ1, bK1, s);
#pragma unroll
      for (int j = 0; j < 8; j++)
        S[(size_t)(kt * 16 + m16) * 16 + j + 8 * lh] = s[j] * 0.125f;
    }
    __syncthreads();

    if (lane < 16) {
      float mx = -3.4e38f;
      for (int k = 0; k < WINW; k++) mx = fmaxf(mx, S[(size_t)k * 16 + lane]);
      float sum = 0.f;
      for (int k = 0; k < WINW; k++) {
        float p = __expf(S[(size_t)k * 16 + lane] - mx);
        S[(size_t)k * 16 + lane] = p;
        sum += p;
      }
      float inv = 1.f / sum;
      for (int k = 0; k < WINW; k++) S[(size_t)k * 16 + lane] *= inv;
    }
    __syncthreads();

    v8f oacc[4] = {{}, {}, {}, {}};
    for (int kc = 0; kc < WINW / 32; kc++) {
      BF16x16 ap;
#pragma unroll
      for (int e = 0; e < 16; e++) {
        int kl = kc * 32 + e + 16 * lh;
        ap.e[e] = (__bf16)S[(size_t)kl * 16 + m16];
      }
#pragma unroll
      for (int nt = 0; nt < 4; nt++) {
        v16bf bV = *(const v16bf*)(Vt + (size_t)(nt * 16 + m16) * WINW + kc * 32 + 16 * lh);
        oacc[nt] = wmma_bf16(ap.v, bV, oacc[nt]);
      }
    }
#pragma unroll
    for (int nt = 0; nt < 4; nt++)
#pragma unroll
      for (int j = 0; j < 8; j++) {
        int p = w * WINW + qt * 16 + j + 8 * lh;
        int r = (p + SHIFTW) & (LL - 1);
        o[(base + r) * CC + h * DD + nt * 16 + m16] = (__bf16)oacc[nt][j];
      }
  }
}

// ---------------- top-k cross attention ----------------
__global__ __launch_bounds__(32) void cross_attn_kernel(
    const __bf16* __restrict__ q, const __bf16* __restrict__ kv,
    __bf16* __restrict__ oc) {
  extern __shared__ float S[];  // [1024][16], key-major
  int blk = blockIdx.x;
  int qt = blk & 255, h = (blk >> 8) & 15, b = blk >> 12;
  int lane = threadIdx.x, lh = lane >> 4, m16 = lane & 15;

  const __bf16* qp = q + ((size_t)(b * LL + qt * 16 + m16)) * CC + h * DD;
  v16bf aQ0 = load_a_ptr(qp);
  v16bf aQ1 = load_a_ptr(qp + 32);

  for (int kt = 0; kt < 64; kt++) {
    const __bf16* kp = kv + ((size_t)(b * 1024 + kt * 16 + m16)) * (2 * CC) + h * DD;
    v16bf b0 = *(const v16bf*)(kp + 16 * lh);
    v16bf b1 = *(const v16bf*)(kp + 32 + 16 * lh);
    v8f s = {};
    s = wmma_bf16(aQ0, b0, s);
    s = wmma_bf16(aQ1, b1, s);
#pragma unroll
    for (int j = 0; j < 8; j++)
      S[(size_t)(kt * 16 + m16) * 16 + j + 8 * lh] = s[j] * 0.125f;
  }
  __syncthreads();

  if (lane < 16) {
    float tv[TOPKK];
    int ti[TOPKK];
#pragma unroll
    for (int j = 0; j < TOPKK; j++) { tv[j] = -3.4e38f; ti[j] = 0; }
    for (int k = 0; k < 1024; k++) {
      float sv = S[(size_t)k * 16 + lane];
      if (sv > tv[TOPKK - 1]) {
        int j = TOPKK - 1;
        while (j > 0 && sv > tv[j - 1]) { tv[j] = tv[j - 1]; ti[j] = ti[j - 1]; j--; }
        tv[j] = sv;
        ti[j] = k;
      }
    }
    float mx = tv[0], sum = 0.f, wgt[TOPKK];
#pragma unroll
    for (int j = 0; j < TOPKK; j++) { wgt[j] = __expf(tv[j] - mx); sum += wgt[j]; }
    float inv = 1.f / sum;
    float acc[DD];
#pragma unroll
    for (int d = 0; d < DD; d++) acc[d] = 0.f;
    for (int j = 0; j < TOPKK; j++) {
      const __bf16* vp = kv + ((size_t)(b * 1024 + ti[j])) * (2 * CC) + CC + h * DD;
      float wj = wgt[j] * inv;
#pragma unroll
      for (int d = 0; d < DD; d++) acc[d] += wj * (float)vp[d];
    }
    __bf16* op = oc + ((size_t)(b * LL + qt * 16 + lane)) * CC + h * DD;
#pragma unroll
    for (int d = 0; d < DD; d++) op[d] = (__bf16)acc[d];
  }
}

// ---------------- host driver ----------------
extern "C" void kernel_launch(void* const* d_in, const int* in_sizes, int n_in,
                              void* d_out, int out_size, void* d_ws, size_t ws_size,
                              hipStream_t stream) {
  const float* x      = (const float*)d_in[0];
  const float* mod    = (const float*)d_in[1];
  const float* ctx    = (const float*)d_in[2];
  const float* w_ada  = (const float*)d_in[3];
  const float* b_ada  = (const float*)d_in[4];
  const float* w_qkv  = (const float*)d_in[5];
  const float* b_qkv  = (const float*)d_in[6];
  const float* w_so   = (const float*)d_in[7];
  const float* b_so   = (const float*)d_in[8];
  const float* w_q    = (const float*)d_in[9];
  const float* b_q    = (const float*)d_in[10];
  const float* w_kv   = (const float*)d_in[11];
  const float* b_kv   = (const float*)d_in[12];
  const float* w_co   = (const float*)d_in[13];
  const float* b_co   = (const float*)d_in[14];
  const float* w_m1   = (const float*)d_in[15];
  const float* b_m1   = (const float*)d_in[16];
  const float* w_m2   = (const float*)d_in[17];
  const float* b_m2   = (const float*)d_in[18];

  const int BN = in_sizes[0] / (LL * CC);
  const int Mrows = BN * LL;
  const int Mctx = BN * 1024;

  char* ws = (char*)d_ws;
  size_t off = 0;
  auto alloc = [&](size_t bytes) {
    size_t p = off;
    off = (off + bytes + 255) & ~(size_t)255;
    return p;
  };

  float*  mbuf   = (float*)(ws + alloc((size_t)BN * 9 * CC * 4));
  __bf16* wqkvb  = (__bf16*)(ws + alloc((size_t)CC * 3 * CC * 2));
  __bf16* wsob   = (__bf16*)(ws + alloc((size_t)CC * CC * 2));
  __bf16* wqb    = (__bf16*)(ws + alloc((size_t)CC * CC * 2));
  __bf16* wkvb   = (__bf16*)(ws + alloc((size_t)CC * 2 * CC * 2));
  __bf16* wcob   = (__bf16*)(ws + alloc((size_t)CC * CC * 2));
  __bf16* wm1b   = (__bf16*)(ws + alloc((size_t)CC * 2 * HIDN * 2));
  __bf16* wm2b   = (__bf16*)(ws + alloc((size_t)HIDN * CC * 2));
  __bf16* ctxb   = (__bf16*)(ws + alloc((size_t)Mctx * CC * 2));
  __bf16* hbuf   = (__bf16*)(ws + alloc((size_t)Mrows * CC * 2));
  float*  x1f    = (float*)(ws + alloc((size_t)Mrows * CC * 4));
  float*  x2f    = (float*)(ws + alloc((size_t)Mrows * CC * 4));
  __bf16* kvb    = (__bf16*)(ws + alloc((size_t)Mctx * 2 * CC * 2));
  // qkvb (3C wide) + oattn (C wide) contiguous; reused: qc -> [0,M*C), ocross ->
  // [M*C,2M*C), g (M x HID) spans qkvb+oattn exactly.
  __bf16* qkvb   = (__bf16*)(ws + alloc((size_t)Mrows * 3 * CC * 2));
  __bf16* oattn  = (__bf16*)(ws + alloc((size_t)Mrows * CC * 2));
  __bf16* qc     = qkvb;
  __bf16* ocross = qkvb + (size_t)Mrows * CC;
  __bf16* gbuf   = qkvb;
  float*  outF   = (float*)d_out;

  mod_gemm_kernel<<<dim3(9 * CC / 128, BN), 128, 0, stream>>>(mod, w_ada, b_ada, mbuf);

  cvt_f32_bf16<<<1024, 256, 0, stream>>>(w_qkv, wqkvb, CC * 3 * CC);
  cvt_f32_bf16<<<1024, 256, 0, stream>>>(w_so,  wsob,  CC * CC);
  cvt_f32_bf16<<<1024, 256, 0, stream>>>(w_q,   wqb,   CC * CC);
  cvt_f32_bf16<<<1024, 256, 0, stream>>>(w_kv,  wkvb,  CC * 2 * CC);
  cvt_f32_bf16<<<1024, 256, 0, stream>>>(w_co,  wcob,  CC * CC);
  cvt_f32_bf16<<<1024, 256, 0, stream>>>(w_m1,  wm1b,  CC * 2 * HIDN);
  cvt_f32_bf16<<<1024, 256, 0, stream>>>(w_m2,  wm2b,  HIDN * CC);
  cvt_f32_bf16<<<1024, 256, 0, stream>>>(ctx,   ctxb,  Mctx * CC);

  // stage 1: windowed self-attention
  ln_mod_kernel<<<Mrows, 256, 0, stream>>>(x, mbuf, 0, 1, hbuf, LL);
  gemm_bf16_kernel<0><<<dim3(3 * CC / 128, Mrows / 128), 256, 0, stream>>>(
      hbuf, wqkvb, b_qkv, qkvb, nullptr, nullptr, nullptr, 0,
      Mrows, 3 * CC, CC, 3 * CC, LL);
  hipFuncSetAttribute((const void*)self_attn_kernel,
                      hipFuncAttributeMaxDynamicSharedMemorySize, 131072);
  self_attn_kernel<<<BN * 128, 64, 131072, stream>>>(qkvb, oattn);
  gemm_bf16_kernel<1><<<dim3(CC / 128, Mrows / 128), 256, 0, stream>>>(
      oattn, wsob, b_so, nullptr, x1f, x, mbuf, 2,
      Mrows, CC, CC, CC, LL);

  // stage 2: top-k cross attention
  ln_mod_kernel<<<Mrows, 256, 0, stream>>>(x1f, mbuf, 3, 4, hbuf, LL);
  gemm_bf16_kernel<0><<<dim3(CC / 128, Mrows / 128), 256, 0, stream>>>(
      hbuf, wqb, b_q, qc, nullptr, nullptr, nullptr, 0,
      Mrows, CC, CC, CC, LL);
  gemm_bf16_kernel<0><<<dim3(2 * CC / 128, Mctx / 128), 256, 0, stream>>>(
      ctxb, wkvb, b_kv, kvb, nullptr, nullptr, nullptr, 0,
      Mctx, 2 * CC, CC, 2 * CC, LL);
  hipFuncSetAttribute((const void*)cross_attn_kernel,
                      hipFuncAttributeMaxDynamicSharedMemorySize, 65536);
  cross_attn_kernel<<<BN * 4096, 32, 65536, stream>>>(qc, kvb, ocross);
  gemm_bf16_kernel<1><<<dim3(CC / 128, Mrows / 128), 256, 0, stream>>>(
      ocross, wcob, b_co, nullptr, x2f, x1f, mbuf, 5,
      Mrows, CC, CC, CC, LL);

  // stage 3: SwiGLU MLP (fused dual-N first GEMM)
  ln_mod_kernel<<<Mrows, 256, 0, stream>>>(x2f, mbuf, 6, 7, hbuf, LL);
  gemm_bf16_kernel<2><<<dim3(HIDN / 64, Mrows / 128), 256, 0, stream>>>(
      hbuf, wm1b, b_m1, gbuf, nullptr, nullptr, nullptr, 0,
      Mrows, HIDN, CC, 2 * HIDN, LL);
  gemm_bf16_kernel<1><<<dim3(CC / 128, Mrows / 128), 256, 0, stream>>>(
      gbuf, wm2b, b_m2, nullptr, outF, x2f, mbuf, 8,
      Mrows, CC, HIDN, CC, LL);
}
